// SurrogateModel_78838419685525
// MI455X (gfx1250) — matrix operand order repaired and checked by reading, not compiled, not run on hardware
//
#include <hip/hip_runtime.h>
#include <hip/hip_bf16.h>

// ---------------------------------------------------------------------------
// CDNA5 (gfx1250) GAT forward. BF16 WMMA GEMMs, TDM (tensor_load_to_lds)
// tile staging, fused attention.
// ---------------------------------------------------------------------------

typedef __attribute__((ext_vector_type(16))) __bf16 v16bf;
typedef __attribute__((ext_vector_type(8)))  __bf16 v8bf;
typedef __attribute__((ext_vector_type(8)))  float  v8f;
typedef __attribute__((ext_vector_type(4)))  unsigned int v4u;
typedef __attribute__((ext_vector_type(8)))  int    v8i;
typedef __attribute__((ext_vector_type(4)))  int    v4i;

typedef unsigned short bfu;   // raw bf16 bits
typedef unsigned int   u32;
typedef unsigned long long u64;

__device__ __forceinline__ bfu f2bf(float f) {
    u32 u = __builtin_bit_cast(u32, f);
    u32 r = u + 0x7FFFu + ((u >> 16) & 1u);       // round-to-nearest-even
    return (bfu)(r >> 16);
}
__device__ __forceinline__ float bf2f(bfu h) {
    u32 u = ((u32)h) << 16;
    return __builtin_bit_cast(float, u);
}

// --------------------------- TDM descriptors --------------------------------
// 2-D tile load (bf16 elements) global -> LDS with a 16B pad after every 64B
// row (=> LDS row stride 80B). D# per CDNA5 ISA §8.3/8.4. Group 1 is
// loop-invariant (tensor_d0 may be any value >= tile_d0; dim0 never OOB).

__device__ __forceinline__ v8i tdm_make_g1(u32 tensor_d0, u32 tensor_d1,
                                           u32 tile_d0, u32 tile_d1,
                                           u64 stride0_elems) {
    v8i g1;
    // data_size=1 (2B) | pad_enable | pad_interval=3 (16 DW) | pad_amount=3 (4 DW)
    g1[0] = (int)((1u << 16) | (1u << 20) | (3u << 22) | (3u << 25));
    g1[1] = (int)((tensor_d0 & 0xFFFFu) << 16);         // dim0[15:0] @ bits 63:48
    g1[2] = (int)((tensor_d0 >> 16) | ((tensor_d1 & 0xFFFFu) << 16));
    g1[3] = (int)((tensor_d1 >> 16) | (tile_d0 << 16)); // tile_dim0 @ 127:112
    g1[4] = (int)(tile_d1);                             // tile_dim1; tile_dim2=0
    g1[5] = (int)(stride0_elems & 0xFFFFFFFFu);         // tensor_dim0_stride lo
    g1[6] = (int)((stride0_elems >> 32) & 0xFFFFu);     // stride hi; stride1=0
    g1[7] = 0;
    return g1;
}

// This toolchain exposes the 6-arg builtin: (g0, g1, g2, g3, g_extra, cpol).
__device__ __forceinline__ void tdm_issue(u32 lds_off, const void* gaddr, v8i g1) {
    u64 ga = (u64)gaddr;
    v4u g0;
    g0.x = 1u;                                            // count=1, user desc
    g0.y = lds_off;                                       // lds_addr [63:32]
    g0.z = (u32)(ga & 0xFFFFFFFFu);                       // global_addr lo
    g0.w = (u32)((ga >> 32) & 0x01FFFFFFu) | 0x80000000u; // addr hi | type=2
    v4i z4 = {0, 0, 0, 0};                                // groups 2/3: 2-D
    v8i z8 = {0, 0, 0, 0, 0, 0, 0, 0};
    __builtin_amdgcn_tensor_load_to_lds(g0, g1, z4, z4, z8, 0);
}

// ---------------------------- prep kernels ---------------------------------

__global__ void cvt_f32_to_bf16(const float* __restrict__ src,
                                bfu* __restrict__ dst, int n) {
    int i = blockIdx.x * blockDim.x + threadIdx.x;
    if (i < n) dst[i] = f2bf(src[i]);
}

// W[k,n] (row-major KxN, fp32) -> Wt[n,k] (row-major NxK, bf16)
__global__ void transpose_w(const float* __restrict__ W,
                            bfu* __restrict__ Wt, int K, int N) {
    int i = blockIdx.x * blockDim.x + threadIdx.x;
    if (i < K * N) {
        int k = i / N, n = i - k * N;
        Wt[(size_t)n * K + k] = f2bf(W[i]);
    }
}

// ------------------------------ GEMM ---------------------------------------
// C[M,N] = A[M,K](bf16) * Wt[N,K](bf16, pre-transposed) -> bf16
// Block: 256 threads = 8 waves. Block tile 128x128, wave tile 32x64 (2x4
// 16x16x32 WMMA tiles). K stepped by 32; tiles staged into double-buffered
// LDS by the Tensor Data Mover (wave 0 issues, TENSORcnt-synced).

#define MT 128
#define NT 128
#define KT 32
#define LDA 40           // padded LDS row stride (elements); 80B, 16B aligned

__global__ __launch_bounds__(256)
void gemm_bf16_wmma(const bfu* __restrict__ X, const bfu* __restrict__ Wt,
                    bfu* __restrict__ H, int M, int K, int N) {
    __shared__ __align__(16) bfu As[2][MT * LDA];
    __shared__ __align__(16) bfu Bs[2][NT * LDA];

    const int tid  = threadIdx.x;
    const int lane = tid & 31;
    const int wave = tid >> 5;
    const int wm   = wave & 3;                 // 4 waves down M
    const int wn   = wave >> 2;                // 2 waves across N
    const int mrow = blockIdx.y * MT;
    const int ncol = blockIdx.x * NT;
    const int l15  = lane & 15;
    const int kh   = lane >> 4;                // K-half select (ISA A/B layout)

    const v8f vzero = {0.f, 0.f, 0.f, 0.f, 0.f, 0.f, 0.f, 0.f};
    v8f acc[2][4];
#pragma unroll
    for (int a = 0; a < 2; ++a)
#pragma unroll
        for (int b = 0; b < 4; ++b) acc[a][b] = vzero;

    const u32 remM = (u32)(M - mrow);          // TDM zero-fills rows >= M
    const u32 remN = (u32)(N - ncol);
    const v8i g1A = tdm_make_g1((u32)K, remM, KT, MT, (u64)K);  // loop-invariant
    const v8i g1B = tdm_make_g1((u32)K, remN, KT, NT, (u64)K);

    // stage K-slab 0: TDM issued by wave 0 only (EXEC-independent, async)
    if (wave == 0) {
        tdm_issue((u32)(size_t)&As[0][0], X + (size_t)mrow * K, g1A);
        tdm_issue((u32)(size_t)&Bs[0][0], Wt + (size_t)ncol * K, g1B);
        __builtin_amdgcn_s_wait_tensorcnt(0);
    }
    __syncthreads();

    const int KS = K / KT;
    for (int ks = 0; ks < KS; ++ks) {
        const int buf = ks & 1;
        if (wave == 0 && ks + 1 < KS) {        // async prefetch next K-slab
            const int k0 = (ks + 1) * KT;
            tdm_issue((u32)(size_t)&As[buf ^ 1][0], X + (size_t)mrow * K + k0, g1A);
            tdm_issue((u32)(size_t)&Bs[buf ^ 1][0], Wt + (size_t)ncol * K + k0, g1B);
        }

        // A fragment: lane = row (lane&15); VGPR0-3 hold K = kh*8..+7,
        // VGPR4-7 hold K = 16+kh*8..+7  (ISA 16-bit 16x32 A layout)
        v16bf afr[2];
#pragma unroll
        for (int tm = 0; tm < 2; ++tm) {
            const bfu* p = &As[buf][(wm * 32 + tm * 16 + l15) * LDA];
            v8bf lo = *(const v8bf*)(p + kh * 8);
            v8bf hi = *(const v8bf*)(p + 16 + kh * 8);
            afr[tm] = __builtin_shufflevector(lo, hi,
                0, 1, 2, 3, 4, 5, 6, 7, 8, 9, 10, 11, 12, 13, 14, 15);
        }
        // B fragment: lane = column; K = kh*8..+7, then 16+kh*8..+7 (same as A)
        v16bf bfr[4];
#pragma unroll
        for (int tn = 0; tn < 4; ++tn) {
            const bfu* p = &Bs[buf][(wn * 64 + tn * 16 + l15) * LDA];
            v8bf lo = *(const v8bf*)(p + kh * 8);
            v8bf hi = *(const v8bf*)(p + 16 + kh * 8);
            bfr[tn] = __builtin_shufflevector(lo, hi,
                0, 1, 2, 3, 4, 5, 6, 7, 8, 9, 10, 11, 12, 13, 14, 15);
        }

#pragma unroll
        for (int tm = 0; tm < 2; ++tm)
#pragma unroll
            for (int tn = 0; tn < 4; ++tn)
                acc[tm][tn] = __builtin_amdgcn_wmma_f32_16x16x32_bf16(
                    false, afr[tm], false, bfr[tn],
                    (short)0, acc[tm][tn], false, false);

        if (wave == 0 && ks + 1 < KS)
            __builtin_amdgcn_s_wait_tensorcnt(0);   // next slab landed in LDS
        __syncthreads();
    }

    // C layout: VGPR r, lanes 0-15 -> M=r, lanes 16-31 -> M=8+r; col = lane&15
    if (mrow + MT <= M) {                      // uniform fast path: no guards
#pragma unroll
        for (int tm = 0; tm < 2; ++tm) {
            const int row0 = mrow + wm * 32 + tm * 16 + kh * 8;
#pragma unroll
            for (int tn = 0; tn < 4; ++tn) {
                const int col = ncol + wn * 64 + tn * 16 + l15;
                bfu* p = H + (size_t)row0 * N + col;
#pragma unroll
                for (int r = 0; r < 8; ++r)
                    p[(size_t)r * N] = f2bf(acc[tm][tn][r]);
            }
        }
    } else {                                   // partial last M-tile
#pragma unroll
        for (int tm = 0; tm < 2; ++tm) {
            const int row0 = mrow + wm * 32 + tm * 16 + kh * 8;
#pragma unroll
            for (int tn = 0; tn < 4; ++tn) {
                const int col = ncol + wn * 64 + tn * 16 + l15;
                bfu* p = H + (size_t)row0 * N + col;
#pragma unroll
                for (int r = 0; r < 8; ++r)
                    if (row0 + r < M) p[(size_t)r * N] = f2bf(acc[tm][tn][r]);
            }
        }
    }
}

// --------------------------- attention -------------------------------------
// One block per graph. Fully-connected 10-node GAT attention + aggregation.
// Reads H (bf16, [10,F]), writes ReLU(alpha@H + b) as bf16 next-layer input.

__global__ __launch_bounds__(256)
void gat_attention(const bfu* __restrict__ Hin, const float* __restrict__ a_src,
                   const float* __restrict__ a_dst, const float* __restrict__ bias,
                   bfu* __restrict__ Xout, int F) {
    __shared__ __align__(16) bfu hs[10 * 1024];
    __shared__ float se[10], sd[10];
    __shared__ float sal[10][10];

    const int b   = blockIdx.x;
    const int tid = threadIdx.x;
    const size_t base = (size_t)b * 10 * F;

    // stage h into LDS (16B chunks, coalesced)
    const int nch = (10 * F) / 8;
    for (int c = tid; c < nch; c += 256)
        *(uint4*)&hs[c * 8] = *(const uint4*)(Hin + base + (size_t)c * 8);

    if (tid < 10) { se[tid] = 0.f; sd[tid] = 0.f; }
    __syncthreads();

    // e_src[n] = h[n]·a_src ; e_dst[n] = h[n]·a_dst  (LDS float atomics)
    float ps[10], pd[10];
#pragma unroll
    for (int n = 0; n < 10; ++n) { ps[n] = 0.f; pd[n] = 0.f; }
    for (int f = tid; f < F; f += 256) {
        float ws = a_src[f], wd = a_dst[f];
#pragma unroll
        for (int n = 0; n < 10; ++n) {
            float h = bf2f(hs[n * F + f]);
            ps[n] += h * ws;
            pd[n] += h * wd;
        }
    }
#pragma unroll
    for (int n = 0; n < 10; ++n) {
        atomicAdd(&se[n], ps[n]);
        atomicAdd(&sd[n], pd[n]);
    }
    __syncthreads();

    // per-destination softmax over leaky_relu(e_src[j] + e_dst[i])
    if (tid < 10) {
        const int i = tid;
        float e[10], m = -3.4e38f;
#pragma unroll
        for (int j = 0; j < 10; ++j) {
            float t = se[j] + sd[i];
            t = (t > 0.f) ? t : 0.2f * t;      // negative_slope = 0.2
            e[j] = t;
            m = fmaxf(m, t);
        }
        float s = 0.f;
#pragma unroll
        for (int j = 0; j < 10; ++j) { e[j] = __expf(e[j] - m); s += e[j]; }
        float inv = 1.f / s;
#pragma unroll
        for (int j = 0; j < 10; ++j) sal[i][j] = e[j] * inv;
    }
    __syncthreads();

    // out[i,f] = relu( sum_j alpha[i][j]*h[j,f] + bias[f] )
    for (int idx = tid; idx < 10 * F; idx += 256) {
        int i = idx / F, f = idx - i * F;
        float acc = bias[f];
#pragma unroll
        for (int j = 0; j < 10; ++j) acc += sal[i][j] * bf2f(hs[j * F + f]);
        acc = fmaxf(acc, 0.f);
        Xout[base + idx] = f2bf(acc);
    }
}

// ------------------------------ heads --------------------------------------
// y[f] = sum_n lin_w[n]*h[n,f] + lin_b ; v = tanh(y)·reg_w + reg_b ;
// c = sigmoid(relu(y)·cls_w + cls_b). One block (128 thr) per graph.

__global__ __launch_bounds__(128)
void heads_kernel(const bfu* __restrict__ X, const float* __restrict__ lin_w,
                  const float* __restrict__ lin_b, const float* __restrict__ reg_w,
                  const float* __restrict__ reg_b, const float* __restrict__ cls_w,
                  const float* __restrict__ cls_b, float* __restrict__ out, int Bn) {
    __shared__ float rv[128], rc[128];
    const int b = blockIdx.x, f = threadIdx.x;
    const size_t base = (size_t)b * 10 * 128;

    float y = lin_b[0];
#pragma unroll
    for (int n = 0; n < 10; ++n) y += lin_w[n] * bf2f(X[base + n * 128 + f]);

    rv[f] = tanhf(y) * reg_w[f];
    rc[f] = fmaxf(y, 0.f) * cls_w[f];
    __syncthreads();
    for (int s = 64; s > 0; s >>= 1) {
        if (f < s) { rv[f] += rv[f + s]; rc[f] += rc[f + s]; }
        __syncthreads();
    }
    if (f == 0) {
        out[b] = rv[0] + reg_b[0];
        float z = rc[0] + cls_b[0];
        out[Bn + b] = 1.f / (1.f + __expf(-z));
    }
}

// ----------------------------- launcher ------------------------------------

extern "C" void kernel_launch(void* const* d_in, const int* in_sizes, int n_in,
                              void* d_out, int out_size, void* d_ws, size_t ws_size,
                              hipStream_t stream) {
    // pytree order: x, edge_index, edge_attr, then params alphabetically:
    //   cls_b, cls_w, convs[0..5]{W,a_src,a_dst,b}, lin_b, lin_w, reg_b, reg_w
    const float* x     = (const float*)d_in[0];
    const float* cls_b = (const float*)d_in[3];
    const float* cls_w = (const float*)d_in[4];
    const float* convW[6], *convAs[6], *convAd[6], *convB[6];
    for (int l = 0; l < 6; ++l) {
        convW[l]  = (const float*)d_in[5 + 4 * l + 0];
        convAs[l] = (const float*)d_in[5 + 4 * l + 1];
        convAd[l] = (const float*)d_in[5 + 4 * l + 2];
        convB[l]  = (const float*)d_in[5 + 4 * l + 3];
    }
    const float* lin_b = (const float*)d_in[29];
    const float* lin_w = (const float*)d_in[30];
    const float* reg_b = (const float*)d_in[31];
    const float* reg_w = (const float*)d_in[32];

    const int Bn = 2000, M = 20000;
    const int Fin[6]  = {128, 512, 512, 1024, 1024, 128};
    const int Fout[6] = {512, 512, 1024, 1024, 128, 128};

    // workspace: buf0 | buf1 (bf16 activations, 20000x1024 each) | Wt (bf16)
    bfu* buf0 = (bfu*)d_ws;
    bfu* buf1 = buf0 + (size_t)M * 1024;
    bfu* wtp  = buf1 + (size_t)M * 1024;
    bfu* wts[6];
    size_t off = 0;
    for (int l = 0; l < 6; ++l) { wts[l] = wtp + off; off += (size_t)Fin[l] * Fout[l]; }

    // prep: x -> bf16 ; W -> bf16 transposed
    const int nx = M * 128;
    hipLaunchKernelGGL(cvt_f32_to_bf16, dim3((nx + 255) / 256), dim3(256), 0, stream,
                       x, buf0, nx);
    for (int l = 0; l < 6; ++l) {
        int kn = Fin[l] * Fout[l];
        hipLaunchKernelGGL(transpose_w, dim3((kn + 255) / 256), dim3(256), 0, stream,
                           convW[l], wts[l], Fin[l], Fout[l]);
    }

    // 6 GATConv layers: GEMM (WMMA + TDM staging) then fused attention
    for (int l = 0; l < 6; ++l) {
        dim3 grid(Fout[l] / 128, (M + 127) / 128);
        hipLaunchKernelGGL(gemm_bf16_wmma, grid, dim3(256), 0, stream,
                           buf0, wts[l], buf1, M, Fin[l], Fout[l]);
        hipLaunchKernelGGL(gat_attention, dim3(Bn), dim3(256), 0, stream,
                           buf1, convAs[l], convAd[l], convB[l], buf0, Fout[l]);
    }

    hipLaunchKernelGGL(heads_kernel, dim3(Bn), dim3(128), 0, stream,
                       buf0, lin_w, lin_b, reg_w, reg_b, cls_w, cls_b,
                       (float*)d_out, Bn);
}
